// AVWGCN_23364622090639
// MI455X (gfx1250) — matrix operand order repaired and checked by reading, not compiled, not run on hardware
//
#include <hip/hip_runtime.h>
#include <hip/hip_bf16.h>

// ---------- types ----------
typedef __bf16 v16bf __attribute__((ext_vector_type(16)));
typedef float  v8f   __attribute__((ext_vector_type(8)));
typedef float  v2f   __attribute__((ext_vector_type(2)));

union ABf { v16bf v; uint4 q[2]; };

static __device__ inline unsigned short f2bf_bits(float f) {
    unsigned u = __builtin_bit_cast(unsigned, f);
    return (unsigned short)((u + 0x7FFFu + ((u >> 16) & 1u)) >> 16);
}

#define NB   4096
#define BB   32
#define CIN  32
#define COUT 32
#define DD   16
#define JW   1024   // B*CIN

// ---------- kernel 1: x [B,N,CIN] f32 -> XT2 [J=1024][M=4096] bf16 ----------
__global__ void xt_kernel(const float* __restrict__ x, unsigned short* __restrict__ xt) {
    size_t o = (size_t)blockIdx.x * 256 + threadIdx.x;   // 4M elements
    int m = (int)(o & 4095);
    int j = (int)(o >> 12);
    int b = j >> 5, c = j & 31;
    xt[o] = f2bf_bits(x[((size_t)b * NB + m) * CIN + c]);
}

// ---------- kernel 2: supports row n = softmax(relu(E[n]·E^T)) -> bf16 ----------
__global__ void supports_kernel(const float* __restrict__ E, unsigned short* __restrict__ S) {
    int n = blockIdx.x;
    int tid = threadIdx.x;                  // 256 threads
    __shared__ float En[16];
    __shared__ float red[256];
    if (tid < 16) En[tid] = E[n * DD + tid];
    __syncthreads();

    float logit[16];
    #pragma unroll
    for (int i = 0; i < 16; i++) {
        int m = tid + i * 256;
        const float* Em = &E[(size_t)m * DD];
        float acc = 0.f;
        #pragma unroll
        for (int d = 0; d < 16; d++) acc += En[d] * Em[d];
        logit[i] = acc > 0.f ? acc : 0.f;   // relu
    }
    float mx = 0.f;
    #pragma unroll
    for (int i = 0; i < 16; i++) mx = fmaxf(mx, logit[i]);
    red[tid] = mx; __syncthreads();
    for (int s = 128; s > 0; s >>= 1) {
        if (tid < s) red[tid] = fmaxf(red[tid], red[tid + s]);
        __syncthreads();
    }
    mx = red[0]; __syncthreads();

    float sum = 0.f;
    #pragma unroll
    for (int i = 0; i < 16; i++) { logit[i] = __expf(logit[i] - mx); sum += logit[i]; }
    red[tid] = sum; __syncthreads();
    for (int s = 128; s > 0; s >>= 1) {
        if (tid < s) red[tid] += red[tid + s];
        __syncthreads();
    }
    float inv = 1.0f / red[0];
    #pragma unroll
    for (int i = 0; i < 16; i++)
        S[(size_t)n * NB + tid + i * 256] = f2bf_bits(logit[i] * inv);
}

// ---------- kernel 3: per-node weights Wg[n][64][32] and bias, 16 nodes/block ----------
__global__ void wgen_kernel(const float* __restrict__ E, const float* __restrict__ Wp,
                            const float* __restrict__ bp, float* __restrict__ Wg,
                            float* __restrict__ biasg) {
    int n0 = blockIdx.x * 16;
    int tid = threadIdx.x;                  // 256 threads
    __shared__ float En[16][16];            // [node][d]
    En[tid >> 4][tid & 15] = E[(size_t)(n0 + (tid >> 4)) * DD + (tid & 15)];
    __syncthreads();

    for (int e = tid; e < 2048; e += 256) { // e = kk*32 + o, kk = k*32+i
        float acc[16];
        #pragma unroll
        for (int ni = 0; ni < 16; ni++) acc[ni] = 0.f;
        #pragma unroll
        for (int d = 0; d < 16; d++) {
            float w = Wp[(size_t)d * 2048 + e];
            #pragma unroll
            for (int ni = 0; ni < 16; ni++) acc[ni] += En[ni][d] * w;
        }
        #pragma unroll
        for (int ni = 0; ni < 16; ni++) Wg[(size_t)(n0 + ni) * 2048 + e] = acc[ni];
    }
    for (int e = tid; e < 512; e += 256) {  // bias: 16 nodes x 32
        int ni = e >> 5, o = e & 31;
        float acc = 0.f;
        #pragma unroll
        for (int d = 0; d < 16; d++) acc += En[ni][d] * bp[d * COUT + o];
        biasg[(size_t)(n0 + ni) * COUT + o] = acc;
    }
}

// ---------- kernel 4: main GEMM  out1[4096x1024] = S[4096x4096]bf16 * XT2^T ----------
// block tile 128(M) x 64(J), 8 waves in 4x2, each wave 2x2 frags of 16x16, K-step 32.
// Double-buffered LDS fed by async global->LDS copies (ASYNCcnt), one barrier/step.
__global__ __launch_bounds__(256) void
gemm_kernel(const unsigned short* __restrict__ A,   // supports bf16 [N][N]
            const unsigned short* __restrict__ Bt,  // XT2 bf16 [J][M] (j-major!)
            float* __restrict__ C) {                // out1 [N][J]
    __shared__ unsigned short As[2][128 * 32];  // [buf][m][k]  8 KB each
    __shared__ unsigned short Bs[2][64 * 32];   // [buf][j][k]  4 KB each
    int tid  = threadIdx.x;
    int lane = tid & 31, wave = tid >> 5;
    int m0 = blockIdx.x * 128, j0 = blockIdx.y * 64;
    int wm = wave >> 1, wn = wave & 1;       // wave tile origin (wm*32, wn*32)
    int Mi = lane & 15, half = lane >> 4;

    // per-thread staging chunk: 16B of row `rowA`, plus a second A chunk 64 rows down
    int rowA = tid >> 2;                     // 0..63
    int offE = (tid & 3) * 8;                // element offset within row
    unsigned gA1 = (unsigned)(((m0 + rowA) * NB + offE) * 2);          // bytes
    unsigned gA2 = gA1 + 64u * NB * 2u;
    unsigned gB  = (unsigned)(((j0 + rowA) * NB + offE) * 2);
    unsigned lA1[2], lA2[2], lB[2];
    #pragma unroll
    for (int p = 0; p < 2; p++) {
        lA1[p] = (unsigned)(size_t)&As[p][rowA * 32 + offE];
        lA2[p] = (unsigned)(size_t)&As[p][(rowA + 64) * 32 + offE];
        lB[p]  = (unsigned)(size_t)&Bs[p][rowA * 32 + offE];
    }

    auto issue = [&](int p, unsigned kbytes) {
        asm volatile(
            "global_load_async_to_lds_b128 %0, %3, %6\n\t"
            "global_load_async_to_lds_b128 %1, %4, %6\n\t"
            "global_load_async_to_lds_b128 %2, %5, %7"
            :
            : "v"(lA1[p]), "v"(lA2[p]), "v"(lB[p]),
              "v"(gA1 + kbytes), "v"(gA2 + kbytes), "v"(gB + kbytes),
              "s"(A), "s"(Bt)
            : "memory");
    };

    v8f acc[2][2];
    #pragma unroll
    for (int i = 0; i < 2; i++)
        #pragma unroll
        for (int j = 0; j < 2; j++) acc[i][j] = v8f{};

    issue(0, 0);                             // prologue prefetch into buf 0

    for (int it = 0; it < 128; ++it) {
        int p = it & 1;
        asm volatile("s_wait_asynccnt 0x0" ::: "memory");  // own copies into buf p done
        __syncthreads();                                   // all waves' copies visible
        if (it + 1 < 128) issue(1 - p, (unsigned)(it + 1) * 64u);  // prefetch next

        const unsigned short* Ap = As[p];
        const unsigned short* Bp = Bs[p];
        ABf afr[2], bfr[2];
        #pragma unroll
        for (int i = 0; i < 2; i++) {
            int r = wm * 32 + i * 16 + Mi;   // A row (local)
            afr[i].q[0] = *(const uint4*)&Ap[r * 32 + half * 8];        // K = half*8 .. +7
            afr[i].q[1] = *(const uint4*)&Ap[r * 32 + 16 + half * 8];   // K = 16+half*8 .. +7
            int jj = wn * 32 + i * 16 + Mi;  // B col (local)
            bfr[i].q[0] = *(const uint4*)&Bp[jj * 32 + half * 8];       // K = half*8 .. +7
            bfr[i].q[1] = *(const uint4*)&Bp[jj * 32 + 16 + half * 8];  // K = 16+half*8 .. +7
        }
        #pragma unroll
        for (int i = 0; i < 2; i++)
            #pragma unroll
            for (int j = 0; j < 2; j++)
                acc[i][j] = __builtin_amdgcn_wmma_f32_16x16x32_bf16(
                    false, afr[i].v, false, bfr[j].v, (short)0, acc[i][j], false, false);
    }
    // store: C/D layout -> VGPR r: row = r + 8*half, col = lane&15
    #pragma unroll
    for (int i = 0; i < 2; i++)
        #pragma unroll
        for (int j = 0; j < 2; j++) {
            int row0 = m0 + wm * 32 + i * 16 + 8 * half;
            int col  = j0 + wn * 32 + j * 16 + Mi;
            #pragma unroll
            for (int r = 0; r < 8; r++)
                C[(size_t)(row0 + r) * JW + col] = acc[i][j][r];
        }
}

// ---------- kernel 5: per-node out[b,n,o] = [x | out1][32x64] * Wn[64x32] + bias ----------
// 1 block (128 thr = 4 waves) per node, each wave one 16x16 frag, f32 WMMA 16x16x4.
// Two accumulators break the serial WMMA D->C chain; full unroll issues all loads early.
__global__ __launch_bounds__(128) void
final_kernel(const float* __restrict__ x, const float* __restrict__ out1,
             const float* __restrict__ Wg, const float* __restrict__ biasg,
             float* __restrict__ out) {
    int n = blockIdx.x;
    __shared__ float Wt[32][66];            // transposed weights [o][kk], padded (8B-aligned rows)
    int tid = threadIdx.x, lane = tid & 31, wave = tid >> 5;

    for (int e = tid; e < 2048; e += 128) { // Wg row-major [kk][o]
        int kk = e >> 5, o = e & 31;
        Wt[o][kk] = Wg[(size_t)n * 2048 + e];
    }
    __syncthreads();

    int Mi = lane & 15, half = lane >> 4;
    int mblk = wave >> 1, nblk = wave & 1;
    int b = mblk * 16 + Mi;                 // A row = batch index
    int o = nblk * 16 + Mi;                 // C col = output channel
    const float* xrow  = &x[((size_t)b * NB + n) * CIN];   // 32 floats (k=0, identity support)
    const float* grow  = &out1[(size_t)n * JW + b * CIN];  // 32 floats (k=1, supports@x)
    v8f acc0 = v8f{}, acc1 = v8f{};

    #pragma unroll
    for (int ks = 0; ks < 16; ks++) {
        int kk = ks * 4 + half * 2;         // lane's K pair (f32 A: VGPR0/1 = K, K+1)
        v2f a  = (ks < 8) ? *(const v2f*)&xrow[kk] : *(const v2f*)&grow[kk - 32];
        v2f bv = *(const v2f*)&Wt[o][kk];
        if (ks & 1)
            acc1 = __builtin_amdgcn_wmma_f32_16x16x4_f32(false, a, false, bv, (short)0, acc1, false, false);
        else
            acc0 = __builtin_amdgcn_wmma_f32_16x16x4_f32(false, a, false, bv, (short)0, acc0, false, false);
    }
    float bias = biasg[(size_t)n * COUT + o];
    #pragma unroll
    for (int r = 0; r < 8; r++) {
        int brow = mblk * 16 + r + 8 * half;
        out[((size_t)brow * NB + n) * COUT + o] = acc0[r] + acc1[r] + bias;
    }
}

// ---------- launcher ----------
extern "C" void kernel_launch(void* const* d_in, const int* in_sizes, int n_in,
                              void* d_out, int out_size, void* d_ws, size_t ws_size,
                              hipStream_t stream) {
    const float* x  = (const float*)d_in[0];   // [32,4096,32]
    const float* Wp = (const float*)d_in[1];   // [16,2,32,32]
    const float* bp = (const float*)d_in[2];   // [16,32]
    const float* E  = (const float*)d_in[3];   // [4096,16]
    float* out = (float*)d_out;                // [32,4096,32]

    char* ws = (char*)d_ws;
    unsigned short* xt    = (unsigned short*)(ws + 0);          //  8 MB  bf16 [1024][4096]
    unsigned short* S     = (unsigned short*)(ws + (8u << 20)); // 32 MB  bf16 [4096][4096]
    float* out1  = (float*)(ws + (40u << 20));                  // 16 MB  f32  [4096][1024]
    float* Wg    = (float*)(ws + (56u << 20));                  // 32 MB  f32  [4096][64][32]
    float* biasg = (float*)(ws + (88u << 20));                  // 0.5 MB f32  [4096][32]

    xt_kernel<<<(JW * NB) / 256, 256, 0, stream>>>(x, xt);
    supports_kernel<<<NB, 256, 0, stream>>>(E, S);
    wgen_kernel<<<NB / 16, 256, 0, stream>>>(E, Wp, bp, Wg, biasg);
    gemm_kernel<<<dim3(NB / 128, JW / 64), 256, 0, stream>>>(S, xt, out1);
    final_kernel<<<NB, 128, 0, stream>>>(x, out1, Wg, biasg, out);
}
